// Attention_38302518346215
// MI455X (gfx1250) — compile-verified, matrix operand order checked by reading, not run on hardware
//
#include <hip/hip_runtime.h>
#include <hip/hip_bf16.h>
#include <stddef.h>
#include <stdint.h>

// ---- problem constants (from reference) ----
#define B_    16
#define NH_   16
#define T_    2048
#define N_    256
#define D_    1024
#define TOPK_ 32

#define COLS  32              // sigma columns owned per workgroup
#define CBLK  (N_ / COLS)     // 8 column-blocks per head
#define INV2PI 0.15915494309189535f
#define TWOPI  6.283185307179586f

typedef __attribute__((ext_vector_type(16))) __bf16 v16bf;
typedef __attribute__((ext_vector_type(8)))  __bf16 v8bf;
typedef __attribute__((ext_vector_type(8)))  float  v8f;
typedef unsigned int u32x4 __attribute__((ext_vector_type(4)));
typedef int          i32x8 __attribute__((ext_vector_type(8)));
typedef int          i32x4 __attribute__((ext_vector_type(4)));

__device__ __forceinline__ v16bf combine16(v8bf lo, v8bf hi) {
  v16bf r;
#pragma unroll
  for (int i = 0; i < 8; ++i) { r[i] = lo[i]; r[i + 8] = hi[i]; }
  return r;
}

__device__ __forceinline__ v8f wmma_bf16(v16bf a, v16bf b, v8f c) {
  return __builtin_amdgcn_wmma_f32_16x16x32_bf16(
      /*neg_a=*/false, a, /*neg_b=*/false, b,
      /*c_mod=*/(short)0, c, /*reuse_a=*/false, /*reuse_b=*/false);
}

// ---------------------------------------------------------------------------
// Tensor Data Mover: DMA one (16 x 256) f32 Q tile (rows strided by NH*T*N)
// from global memory into LDS.  D# bit layout per CDNA5 ISA ch. 8.
// Descriptor groups are uniform -> land in SGPRs as VIMAGE expects.
// This toolchain exposes the 6-arg builtin (g0, g1, g2, g3, g4, cpol);
// groups 2+ are unused for a 2D tensor.
// ---------------------------------------------------------------------------
__device__ __forceinline__ void tdm_load_q_tile(const float* gptr, unsigned lds_off) {
  const unsigned long long ga = (unsigned long long)(uintptr_t)gptr;
  u32x4 g0;
  g0[0] = 1u;                                           // count=1, is_restore=0
  g0[1] = lds_off;                                      // lds_addr (bytes)
  g0[2] = (unsigned)(ga & 0xFFFFFFFFu);                 // global_addr[31:0]
  g0[3] = (unsigned)((ga >> 32) & 0x1FFFFFFu)           // global_addr[56:32]
          | (2u << 30);                                 // type = 2 ("image")
  const unsigned long long s0 = (unsigned long long)NH_ * T_ * N_;  // row stride (elems)
  i32x8 g1;
  g1[0] = (int)(2u << 16);                   // wg_mask=0, data_size=2 (4 bytes)
  g1[1] = (int)((N_ & 0xFFFFu) << 16);       // tensor_dim0[15:0] = 256
  g1[2] = (int)((B_ & 0xFFFFu) << 16);       // dim0[31:16]=0 | tensor_dim1[15:0] = 16
  g1[3] = (int)((unsigned)N_ << 16);         // dim1[31:16]=0 | tile_dim0 = 256
  g1[4] = (int)(B_ & 0xFFFFu);               // tile_dim1 = 16, tile_dim2 = 0
  g1[5] = (int)(s0 & 0xFFFFFFFFu);           // tensor_dim0_stride[31:0]
  g1[6] = (int)((s0 >> 32) & 0xFFFFu);       // stride[47:32] | dim1_stride[15:0]=0
  g1[7] = 0;
  i32x4 z4; z4[0] = 0; z4[1] = 0; z4[2] = 0; z4[3] = 0;
  i32x8 z8;
#pragma unroll
  for (int i = 0; i < 8; ++i) z8[i] = 0;
  __builtin_amdgcn_tensor_load_to_lds(g0, g1, z4, z4, z8, 0);
}

// ---------------------------------------------------------------------------
// Kernel 0: zero Y_agg accumulator in workspace.
// ---------------------------------------------------------------------------
__global__ __launch_bounds__(256) void zero_kernel(float4* __restrict__ p, int n4) {
  int i = blockIdx.x * 256 + threadIdx.x;
  if (i < n4) p[i] = make_float4(0.f, 0.f, 0.f, 0.f);
}

// ---------------------------------------------------------------------------
// Kernel 1: precompute the global activity gate do[t] = mean(rope(Q)_t > 0) <= 0.3
// Depends only on Q -> fully parallel over t; removes all cross-block sync
// from the recurrent scan.  One block per t; thread (b,h) scans one row.
// ---------------------------------------------------------------------------
__global__ __launch_bounds__(256) void flag_kernel(const float* __restrict__ Q,
                                                   int* __restrict__ doflag) {
  const int t = blockIdx.x;
  const int tid = threadIdx.x;
  const int b = tid >> 4, h = tid & 15;
  const float* qrow = Q + (((size_t)(b * NH_ + h)) * T_ + t) * N_;
  const float tf = (float)t;
  int cnt = 0;
#pragma unroll 4
  for (int p = 0; p < 128; ++p) {
    float2 v = *(const float2*)&qrow[2 * p];
    float fr = exp2f(-(float)(2 * p) * (1.0f / 16.0f)) * INV2PI;
    float ph = tf * fr;
    ph = (ph - floorf(ph)) * TWOPI;
    float s, c;
    __sincosf(ph, &s, &c);
    cnt += ((v.x * c - v.y * s) > 0.f) ? 1 : 0;
    cnt += ((v.y * c + v.x * s) > 0.f) ? 1 : 0;
  }
#pragma unroll
  for (int off = 16; off; off >>= 1) cnt += __shfl_xor(cnt, off, 32);
  __shared__ int red[8];
  if ((tid & 31) == 0) red[tid >> 5] = cnt;
  __syncthreads();
  if (tid == 0) {
    int tot = 0;
#pragma unroll
    for (int w = 0; w < 8; ++w) tot += red[w];
    doflag[t] = ((float)tot * (1.0f / (B_ * NH_ * N_)) <= 0.3f) ? 1 : 0;
  }
}

// ---------------------------------------------------------------------------
// Kernel 2: the recurrent scan.
// grid = NH_*CBLK workgroups; block (h, cb) owns sigma^T columns
// [cb*COLS, cb*COLS+COLS) of head h, resident in LDS (bf16) for all T steps.
// TDM prefetches step t+1's Q tile into LDS while step t computes.
// ---------------------------------------------------------------------------
__global__ __launch_bounds__(256) void scan_kernel(const float* __restrict__ Q,
                                                   float* __restrict__ Yagg,
                                                   const int* __restrict__ doflag) {
  __shared__ __align__(16) __bf16 sigmaB[COLS][N_];     // sigma^T block (16 KB)
  __shared__ __align__(16) __bf16 XA[B_][N_];           // RoPE'd x, A layout (8 KB)
  __shared__ __align__(16) __bf16 ST[N_][16];           // sparse S^T (8 KB)
  __shared__ __align__(8)  unsigned char Hc[COLS][N_];  // Hebb counters (8 KB)
  __shared__ __align__(16) float QS[B_][N_];            // TDM Q staging (16 KB)

  const int tid  = threadIdx.x;
  const int lane = tid & 31;
  const int wave = tid >> 5;           // 0..7
  const int h     = blockIdx.x / CBLK;
  const int cbase = (blockIdx.x % CBLK) * COLS;
  const int lo8   = (lane >> 4) * 8;   // K sub-offset per 16-bit WMMA layout
  const int hi    = lane >> 4;
  const int lrow  = lane & 15;

  // zero state
  {
    __bf16* sp = &sigmaB[0][0];
    unsigned char* hp = &Hc[0][0];
    for (int i = tid; i < COLS * N_; i += 256) { sp[i] = (__bf16)0.0f; hp[i] = 0; }
  }

  // phase-A geometry: thread owns 16 consecutive n's of one batch row
  const int ab = tid >> 4;                  // batch row 0..15
  const int an0 = (tid & 15) * 16;          // n offset
  float fr[8];
#pragma unroll
  for (int p = 0; p < 8; ++p)
    fr[p] = exp2f(-(float)(an0 + 2 * p) * (1.0f / 16.0f)) * INV2PI;

  const unsigned qs_off = (unsigned)(uintptr_t)&QS[0][0];  // LDS byte offset
  const float* qhead = Q + ((size_t)h * T_) * N_;          // tile base for b=0

  if (wave == 0) tdm_load_q_tile(qhead, qs_off);           // prefetch t=0

  for (int t = 0; t < T_; ++t) {
    // ---- (a) wait Q tile DMA; also closes previous step's sigma update ----
    if (wave == 0) __builtin_amdgcn_s_wait_tensorcnt(0);
    __syncthreads();

    // ---- phase A: RoPE from LDS staging -> XA (bf16) ----
    const float tf = (float)t;
    float v[16];
#pragma unroll
    for (int q4 = 0; q4 < 4; ++q4) {
      float4 f = *(const float4*)&QS[ab][an0 + q4 * 4];
      v[q4 * 4 + 0] = f.x; v[q4 * 4 + 1] = f.y; v[q4 * 4 + 2] = f.z; v[q4 * 4 + 3] = f.w;
    }
#pragma unroll
    for (int p = 0; p < 8; ++p) {
      float ph = tf * fr[p];
      ph = (ph - floorf(ph)) * TWOPI;
      float s, c;
      __sincosf(ph, &s, &c);
      float e = v[2 * p], o = v[2 * p + 1];
      XA[ab][an0 + 2 * p]     = (__bf16)(e * c - o * s);
      XA[ab][an0 + 2 * p + 1] = (__bf16)(o * c + e * s);
    }
    __syncthreads();   // (b) XA ready; QS consumed

    // ---- TDM prefetch of step t+1 (wave 0; uniform descriptors -> SGPRs) ----
    if (wave == 0 && t + 1 < T_) tdm_load_q_tile(qhead + (size_t)(t + 1) * N_, qs_off);

    // ---- phase Y: y = X @ sigma (waves 0..1, one 16x16 col-tile each) ----
    v8f acc = {};
    if (wave < 2) {
      const int mcol = wave * 16 + lrow;    // local sigma^T row = output column
#pragma unroll
      for (int k0 = 0; k0 < N_; k0 += 32) {
        v16bf a = combine16(*(const v8bf*)&XA[lrow][k0 + lo8],
                            *(const v8bf*)&XA[lrow][k0 + 16 + lo8]);
        v16bf b = combine16(*(const v8bf*)&sigmaB[mcol][k0 + lo8],
                            *(const v8bf*)&sigmaB[mcol][k0 + 16 + lo8]);
        acc = wmma_bf16(a, b, acc);
      }
    }

    // ---- phase B: top-32 per batch row (2 rows per wave) -> ST ----
#pragma unroll
    for (int rr = 0; rr < 2; ++rr) {
      const int row = wave * 2 + rr;
      // each lane owns 8 contiguous bf16 -> single ds_load_b128
      v8bf xv = *(const v8bf*)&XA[row][lane * 8];
      float lv[8];
#pragma unroll
      for (int j = 0; j < 8; ++j) lv[j] = (float)xv[j];
      unsigned cons = 0;
      float selv = 0.f; int seli = 0;
      for (int it = 0; it < TOPK_; ++it) {
        float m = -3.0e38f; int gi = 0x7fffffff;
#pragma unroll
        for (int j = 0; j < 8; ++j)
          if (!(cons & (1u << j)) && lv[j] > m) { m = lv[j]; gi = lane * 8 + j; }
#pragma unroll
        for (int off = 16; off; off >>= 1) {
          float om = __shfl_xor(m, off, 32);
          int  ogi = __shfl_xor(gi, off, 32);
          if (om > m || (om == m && ogi < gi)) { m = om; gi = ogi; }
        }
        if ((gi >> 3) == lane) cons |= 1u << (gi & 7);
        if (lane == it) { selv = m; seli = gi; }
      }
#pragma unroll
      for (int j = 0; j < 8; ++j) ST[lane * 8 + j][row] = (__bf16)0.0f;
      ST[seli][row] = (__bf16)selv;   // lane i scatters the i-th largest
    }
    __syncthreads();   // (c)

    // ---- y head-reduction into HBM (pre-update sigma, per reference) ----
    if (wave < 2) {
      const int mg = cbase + wave * 16 + lrow;
#pragma unroll
      for (int i = 0; i < 8; ++i) {
        int bb = i + 8 * hi;
        atomicAdd(&Yagg[((size_t)bb * T_ + t) * N_ + mg], acc[i]);
      }
    }

    // ---- phase C: Hebbian update (gated; uniform branch) ----
    if (doflag[t]) {
#pragma unroll
      for (int tt = 0; tt < 4; ++tt) {
        const int tile = wave * 4 + tt;        // 32 tiles of 16x16 over [COLS x N_]
        const int tm = (tile & 1) * 16;        // col-tile within block
        const int tn = (tile >> 1) * 16;       // n-tile
        // hebb tile = (S^T S)[tn:, cbase+tm:] ; K = b in 0..15, padded to 32
        v8bf z8 = {};
        v16bf a = combine16(*(const v8bf*)&ST[tn + lrow][lo8], z8);
        v16bf b = combine16(*(const v8bf*)&ST[cbase + tm + lrow][lo8], z8);
        v8f hb = {};
        hb = wmma_bf16(a, b, hb);
        // vectorized RMW: 8 contiguous sigma elems (b128) + 8 H bytes (uint2)
        const int m  = tm + lrow;
        const int nb = tn + 8 * hi;
        v8bf sv = *(const v8bf*)&sigmaB[m][nb];
        uint2 hv = *(const uint2*)&Hc[m][nb];
        v8bf ns;
        unsigned nh0 = 0, nh1 = 0;
#pragma unroll
        for (int i = 0; i < 8; ++i) {
          unsigned hcv = ((i < 4 ? hv.x >> (8 * i) : hv.y >> (8 * (i - 4))) & 0xFFu);
          float sg  = (float)sv[i];
          float lam = 0.01f * __expf(-0.1f * (float)hcv);
          float hbv = hb[i];
          ns[i] = (__bf16)fmaxf(sg + 0.05f * hbv - lam * sg, 0.0f);
          if (hbv > 0.0f && hcv < 255u) hcv++;
          if (i < 4) nh0 |= hcv << (8 * i); else nh1 |= hcv << (8 * (i - 4));
        }
        *(v8bf*)&sigmaB[m][nb] = ns;
        uint2 nhv; nhv.x = nh0; nhv.y = nh1;
        *(uint2*)&Hc[m][nb] = nhv;
      }
    }
  }
}

// ---------------------------------------------------------------------------
// Kernel 3: out = Y_agg (32768x256) @ W_out^T (256x1024), bf16 WMMA GEMM.
// 128x128 output tile per block, K staged through LDS in chunks of 64,
// next chunk prefetched into L2 via global_prefetch.
// ---------------------------------------------------------------------------
__global__ __launch_bounds__(256) void proj_kernel(const float* __restrict__ Yagg,
                                                   const float* __restrict__ W,
                                                   float* __restrict__ out) {
  __shared__ __align__(16) __bf16 Abuf[128][64];
  __shared__ __align__(16) __bf16 Bbuf[128][64];
  const int tid  = threadIdx.x;
  const int lane = tid & 31;
  const int wave = tid >> 5;
  const int lrow = lane & 15;
  const int lo8  = (lane >> 4) * 8;
  const int mbase = blockIdx.x * 128;   // over B*T = 32768
  const int nbase = blockIdx.y * 128;   // over D = 1024

  v8f acc[8];
#pragma unroll
  for (int j = 0; j < 8; ++j) acc[j] = (v8f){};

  for (int kc = 0; kc < N_; kc += 64) {
    // stage A (f32 -> bf16) and B tiles; 2048 float4s each, 256 threads
    for (int idx = tid; idx < 128 * 16; idx += 256) {
      int r = idx >> 4, c4 = idx & 15;
      float4 f = *(const float4*)&Yagg[(size_t)(mbase + r) * N_ + kc + c4 * 4];
      __bf16* d = &Abuf[r][c4 * 4];
      d[0] = (__bf16)f.x; d[1] = (__bf16)f.y; d[2] = (__bf16)f.z; d[3] = (__bf16)f.w;
    }
    for (int idx = tid; idx < 128 * 16; idx += 256) {
      int r = idx >> 4, c4 = idx & 15;
      float4 f = *(const float4*)&W[(size_t)(nbase + r) * N_ + kc + c4 * 4];
      __bf16* d = &Bbuf[r][c4 * 4];
      d[0] = (__bf16)f.x; d[1] = (__bf16)f.y; d[2] = (__bf16)f.z; d[3] = (__bf16)f.w;
    }
    if (kc + 64 < N_) {   // L2 prefetch of next K chunk (global_prefetch_b8)
      int r = tid >> 1, half = (tid & 1) * 32;
      __builtin_prefetch(&Yagg[(size_t)(mbase + r) * N_ + kc + 64 + half], 0, 1);
      __builtin_prefetch(&W[(size_t)(nbase + r) * N_ + kc + 64 + half], 0, 1);
    }
    __syncthreads();
#pragma unroll
    for (int ks = 0; ks < 64; ks += 32) {
      v16bf a = combine16(*(const v8bf*)&Abuf[wave * 16 + lrow][ks + lo8],
                          *(const v8bf*)&Abuf[wave * 16 + lrow][ks + 16 + lo8]);
#pragma unroll
      for (int tj = 0; tj < 8; ++tj) {
        v16bf b = combine16(*(const v8bf*)&Bbuf[tj * 16 + lrow][ks + lo8],
                            *(const v8bf*)&Bbuf[tj * 16 + lrow][ks + 16 + lo8]);
        acc[tj] = wmma_bf16(a, b, acc[tj]);
      }
    }
    __syncthreads();
  }

#pragma unroll
  for (int tj = 0; tj < 8; ++tj) {
#pragma unroll
    for (int i = 0; i < 8; ++i) {
      int m = mbase + wave * 16 + i + 8 * (lane >> 4);
      int d = nbase + tj * 16 + lrow;
      out[(size_t)m * D_ + d] = acc[tj][i];
    }
  }
}

// ---------------------------------------------------------------------------
extern "C" void kernel_launch(void* const* d_in, const int* in_sizes, int n_in,
                              void* d_out, int out_size, void* d_ws, size_t ws_size,
                              hipStream_t stream) {
  const float* Q = (const float*)d_in[0];   // (B, nh, T, N) f32
  // d_in[1] = K (== Q), d_in[2] = V (unused by reference)
  const float* W = (const float*)d_in[3];   // (D, N) f32
  float* out  = (float*)d_out;              // (B, 1, T, D) f32
  float* Yagg = (float*)d_ws;               // (B, T, N) f32 accumulator
  int* doflag = (int*)((char*)d_ws + (size_t)B_ * T_ * N_ * sizeof(float));

  const int n4 = (B_ * T_ * N_) / 4;
  zero_kernel<<<(n4 + 255) / 256, 256, 0, stream>>>((float4*)Yagg, n4);
  flag_kernel<<<T_, 256, 0, stream>>>(Q, doflag);
  scan_kernel<<<NH_ * CBLK, 256, 0, stream>>>(Q, Yagg, doflag);
  proj_kernel<<<dim3((B_ * T_) / 128, D_ / 128), 256, 0, stream>>>(Yagg, W, out);
}
